// MemoAttention_2491081032426
// MI455X (gfx1250) — matrix operand order repaired and checked by reading, not compiled, not run on hardware
//
#include <hip/hip_runtime.h>
#include <hip/hip_bf16.h>

#define DM 1024
#define NH 16
#define HD 64
#define BB 2
#define SS 2048

typedef __attribute__((ext_vector_type(16))) __bf16         v16bf;
typedef __attribute__((ext_vector_type(8)))  float          v8f;
typedef __attribute__((ext_vector_type(4)))  unsigned int   u32x4;
typedef __attribute__((ext_vector_type(8)))  unsigned int   u32x8;

__device__ __forceinline__ float bf2f(unsigned short u) {
  unsigned int x = ((unsigned int)u) << 16;
  return __builtin_bit_cast(float, x);
}
__device__ __forceinline__ unsigned short f2bf(float f) {
  unsigned int x = __builtin_bit_cast(unsigned int, f);
  x += 0x7fffu + ((x >> 16) & 1u);   // round-to-nearest-even
  return (unsigned short)(x >> 16);
}
__device__ __forceinline__ v16bf mkfrag(u32x4 a, u32x4 b) {
  u32x8 r = __builtin_shufflevector(a, b, 0, 1, 2, 3, 4, 5, 6, 7);
  return __builtin_bit_cast(v16bf, r);
}
__device__ __forceinline__ v8f wmma_bf(v16bf a, v16bf b, v8f c) {
  return __builtin_amdgcn_wmma_f32_16x16x32_bf16(false, a, false, b, (short)0, c,
                                                 false, false);
}
__device__ __forceinline__ float sigmoidf(float x) { return 1.f / (1.f + __expf(-x)); }

// ------------------------------------------------ fp32[K][N] -> bf16 W^T[N][K]
__global__ __launch_bounds__(256) void cvtT_k(const float* __restrict__ s,
                                              unsigned short* __restrict__ d) {
  const int i = blockIdx.x * 256 + threadIdx.x;  // 262144 threads, 4 elems each
  const int k = i >> 8;
  const int n0 = (i & 255) * 4;
  const float* p = s + (size_t)k * DM + n0;
#pragma unroll
  for (int j = 0; j < 4; j++) d[(size_t)(n0 + j) * DM + k] = f2bf(p[j]);
}

// ---------------------------------------------------------------- tiled WMMA GEMM
// C[M=4096,N=1024] = A[M,K=1024] @ W[K,N] with W given transposed (Wt[N][K] bf16).
// Block tile 64x128, 8 waves, wave = 16 rows x 64 cols (4 frags), K-step 32.
// MODE 0: A=f32 input, out bf16 head-major   [b,h,s,d]   (Q,K)
// MODE 3: A=f32 input, out bf16 head-transp. [b,h,d,s]   (V)
// MODE 1: A=bf16, out bf16 = aux * sigmoid(C + bias)     (gate)
// MODE 2: A=bf16, out f32 row-major                      (final projection)
template <int MODE>
__global__ __launch_bounds__(256) void gemm_k(
    const float* __restrict__ Af, const unsigned short* __restrict__ Ab,
    const unsigned short* __restrict__ Wt, const float* __restrict__ bias,
    const unsigned short* __restrict__ aux, unsigned short* __restrict__ Db,
    float* __restrict__ Df) {
  __shared__ unsigned short As[64 * 48];    // [m][k], row stride 48 (96B)
  __shared__ unsigned short Bs[128 * 48];   // [n][k], row stride 48

  const int bm = blockIdx.x * 64;
  const int bn = blockIdx.y * 128;
  const int tid = threadIdx.x;
  const int lane = tid & 31;
  const int wid = tid >> 5;
  const int wr = wid & 3;    // wave row strip (x16)
  const int wc = wid >> 2;   // wave col half (x64)
  const int hl = lane >> 4;
  const int ln = lane & 15;

  v8f acc[4];
  for (int j = 0; j < 4; j++)
    for (int i = 0; i < 8; i++) acc[j][i] = 0.f;

  const int ar = tid >> 2;          // 0..63
  const int ac = (tid & 3) * 8;     // 0,8,16,24
  const int bnr = tid >> 1;         // 0..127
  const int bkc = (tid & 1) * 16;   // 0,16

  for (int kt = 0; kt < DM; kt += 32) {
    // ---- stage A tile (convert to bf16 if from f32 input)
    {
      unsigned short* d = &As[ar * 48 + ac];
      if constexpr (MODE == 0 || MODE == 3) {
        const float* p = Af + (size_t)(bm + ar) * DM + kt + ac;
        u32x4 w0;
#pragma unroll
        for (int i = 0; i < 4; i++) {
          unsigned short l0 = f2bf(p[2 * i]);
          unsigned short h0 = f2bf(p[2 * i + 1]);
          w0[i] = (unsigned int)l0 | ((unsigned int)h0 << 16);
        }
        *(u32x4*)d = w0;
      } else {
        *(u32x4*)d = *(const u32x4*)(Ab + (size_t)(bm + ar) * DM + kt + ac);
      }
    }
    // ---- stage Wt tile: already [n][k] -> contiguous b128 copies
    {
      const unsigned short* p = Wt + (size_t)(bn + bnr) * DM + kt + bkc;
      unsigned short* d = &Bs[bnr * 48 + bkc];
      *(u32x4*)d = *(const u32x4*)p;
      *(u32x4*)(d + 8) = *(const u32x4*)(p + 8);
    }
    __syncthreads();
    // ---- WMMA
    {
      const int am = wr * 16 + ln;
      const int a0 = hl * 8;
      v16bf af = mkfrag(*(const u32x4*)&As[am * 48 + a0],
                        *(const u32x4*)&As[am * 48 + a0 + 16]);
#pragma unroll
      for (int j = 0; j < 4; j++) {
        const int bc = wc * 64 + j * 16 + ln;
        const int kb = hl * 8;
        v16bf bfr = mkfrag(*(const u32x4*)&Bs[bc * 48 + kb],
                           *(const u32x4*)&Bs[bc * 48 + kb + 16]);
        acc[j] = wmma_bf(af, bfr, acc[j]);
      }
    }
    __syncthreads();
  }

  // ---- epilogue
#pragma unroll
  for (int j = 0; j < 4; j++) {
    const int col = bn + wc * 64 + j * 16 + ln;
#pragma unroll
    for (int r = 0; r < 8; r++) {
      const int row = bm + wr * 16 + r + 8 * hl;
      float v = acc[j][r];
      if constexpr (MODE == 0) {
        const int b = row >> 11;
        const int s = row & (SS - 1);
        const int hh = col >> 6;
        const int d = col & 63;
        Db[(((size_t)(b * NH + hh)) * SS + s) * HD + d] = f2bf(v);
      } else if constexpr (MODE == 3) {
        const int b = row >> 11;
        const int s = row & (SS - 1);
        const int hh = col >> 6;
        const int d = col & 63;
        Db[(((size_t)(b * NH + hh)) * HD + d) * SS + s] = f2bf(v);
      } else if constexpr (MODE == 1) {
        const float g = sigmoidf(v + bias[col]);
        const float a = bf2f(aux[(size_t)row * DM + col]);
        Db[(size_t)row * DM + col] = f2bf(a * g);
      } else {
        Df[(size_t)row * DM + col] = v;
      }
    }
  }
}

// ---------------------------------------------------------------- flash attention
// 1 block per (b,h,64-query tile); 4 waves; wave owns 16 query rows.
// Q/K head-major [b,h,s,d]; V head-transposed [b,h,d,s] -> all fragment loads
// are contiguous global b128; only LDS use is the per-wave P layout round-trip.
__global__ __launch_bounds__(128) void attn_k(const unsigned short* __restrict__ Qh,
                                              const unsigned short* __restrict__ Kh,
                                              const unsigned short* __restrict__ Vh,
                                              const float* __restrict__ sw,
                                              unsigned short* __restrict__ O) {
  __shared__ float gS[64];                  // per-row gate*invsqrt multiplier
  __shared__ unsigned short Pb[4][16 * 64]; // per-wave P (probs) [row][key]

  const int bid = blockIdx.x;
  const int qt = bid & 31;
  const int hh = (bid >> 5) & (NH - 1);
  const int bb = bid >> 9;
  const size_t hb = ((size_t)(bb * NH + hh)) * SS * HD;
  const unsigned short* Q = Qh + hb;
  const unsigned short* K = Kh + hb;
  const unsigned short* V = Vh + hb;   // [d][s] within head

  const int tid = threadIdx.x;
  const int wid = tid >> 5;
  const int lane = tid & 31;
  const int hl = lane >> 4;
  const int ln = lane & 15;

  // per-query sigmoid gate * 2 * (1/sqrt(64)) folded into one multiplier
  if (tid < 64) {
    const unsigned short* qr = Q + (size_t)(qt * 64 + tid) * HD;
    const float* s = sw + hh * HD;
    float dot = 0.f;
    for (int d = 0; d < HD; d++) dot += bf2f(qr[d]) * s[d];
    gS[tid] = (2.f * sigmoidf(dot)) * 0.125f;
  }
  __syncthreads();

  float g8[8];
#pragma unroll
  for (int r = 0; r < 8; r++) g8[r] = gS[wid * 16 + r + 8 * hl];

  // Q A-fragments resident for the whole kernel
  const int qrow = qt * 64 + wid * 16 + ln;
  const unsigned short* qp = Q + (size_t)qrow * HD;
  v16bf qa[2];
#pragma unroll
  for (int s = 0; s < 2; s++) {
    const int a0 = 8 * hl + 32 * s;
    qa[s] = mkfrag(*(const u32x4*)(qp + a0), *(const u32x4*)(qp + a0 + 16));
  }

  float mrow[8], lrow[8];
  v8f o[4];
#pragma unroll
  for (int r = 0; r < 8; r++) { mrow[r] = -1e30f; lrow[r] = 0.f; }
  for (int j = 0; j < 4; j++)
    for (int i = 0; i < 8; i++) o[j][i] = 0.f;

  for (int kt = 0; kt < SS / 64; kt++) {
    // scores: Q(16x64) @ K^T(64x64); K B-frags direct from global
    v8f sc[4];
#pragma unroll
    for (int j = 0; j < 4; j++) {
      const unsigned short* kp = K + (size_t)(kt * 64 + j * 16 + ln) * HD;
      __builtin_prefetch(kp + 64 * HD, 0, 3);   // next key tile
      v8f z;
      for (int i = 0; i < 8; i++) z[i] = 0.f;
#pragma unroll
      for (int st = 0; st < 2; st++) {
        const int k0 = 8 * hl + 32 * st;
        v16bf kb = mkfrag(*(const u32x4*)(kp + k0), *(const u32x4*)(kp + k0 + 16));
        z = wmma_bf(qa[st], kb, z);
      }
      sc[j] = z;
    }
    // per-row gate multiplier, online softmax (rows live in lane halves)
    for (int j = 0; j < 4; j++)
      for (int r = 0; r < 8; r++) sc[j][r] *= g8[r];
    float mt[8];
#pragma unroll
    for (int r = 0; r < 8; r++) {
      float v = sc[0][r];
      for (int j = 1; j < 4; j++) v = fmaxf(v, sc[j][r]);
      mt[r] = v;
    }
    for (int m = 1; m < 16; m <<= 1)
#pragma unroll
      for (int r = 0; r < 8; r++) mt[r] = fmaxf(mt[r], __shfl_xor(mt[r], m, 32));
    float alpha[8];
#pragma unroll
    for (int r = 0; r < 8; r++) {
      float mn = fmaxf(mrow[r], mt[r]);
      alpha[r] = __expf(mrow[r] - mn);
      mrow[r] = mn;
    }
    float rs[8];
    for (int r = 0; r < 8; r++) rs[r] = 0.f;
#pragma unroll
    for (int j = 0; j < 4; j++)
#pragma unroll
      for (int r = 0; r < 8; r++) {
        float p = __expf(sc[j][r] - mrow[r]);
        rs[r] += p;
        Pb[wid][(r + 8 * hl) * 64 + j * 16 + ln] = f2bf(p);
      }
    for (int m = 1; m < 16; m <<= 1)
#pragma unroll
      for (int r = 0; r < 8; r++) rs[r] += __shfl_xor(rs[r], m, 32);
#pragma unroll
    for (int r = 0; r < 8; r++) lrow[r] = lrow[r] * alpha[r] + rs[r];
    for (int j = 0; j < 4; j++)
      for (int r = 0; r < 8; r++) o[j][r] *= alpha[r];

    // O += P(16x64) @ V(64x64); P via per-wave LDS, V direct from global [d][s]
    const unsigned short* pp = &Pb[wid][ln * 64];
#pragma unroll
    for (int st = 0; st < 2; st++) {
      const int a0 = 8 * hl + 32 * st;
      v16bf pa = mkfrag(*(const u32x4*)(pp + a0), *(const u32x4*)(pp + a0 + 16));
#pragma unroll
      for (int j = 0; j < 4; j++) {
        const unsigned short* vp =
            V + (size_t)(j * 16 + ln) * SS + kt * 64 + 8 * hl + 32 * st;
        if (st == 0) __builtin_prefetch(vp + 64, 0, 3);  // next key tile
        v16bf vb = mkfrag(*(const u32x4*)(vp), *(const u32x4*)(vp + 16));
        o[j] = wmma_bf(pa, vb, o[j]);
      }
    }
  }

  float il[8];
#pragma unroll
  for (int r = 0; r < 8; r++) il[r] = 1.f / lrow[r];
#pragma unroll
  for (int j = 0; j < 4; j++) {
    const int col = hh * HD + j * 16 + ln;
#pragma unroll
    for (int r = 0; r < 8; r++) {
      const int row = qt * 64 + wid * 16 + r + 8 * hl;
      O[((size_t)(bb * SS + row)) * DM + col] = f2bf(o[j][r] * il[r]);
    }
  }
}

// ----------------------------------------------------------------
extern "C" void kernel_launch(void* const* d_in, const int* in_sizes, int n_in,
                              void* d_out, int out_size, void* d_ws, size_t ws_size,
                              hipStream_t stream) {
  const float* q = (const float*)d_in[0];
  const float* k = (const float*)d_in[1];
  const float* v = (const float*)d_in[2];
  const float* wq = (const float*)d_in[3];
  const float* wk = (const float*)d_in[4];
  const float* wv = (const float*)d_in[5];
  const float* wo = (const float*)d_in[6];
  const float* wg = (const float*)d_in[7];
  const float* gb = (const float*)d_in[8];
  const float* swt = (const float*)d_in[9];

  const size_t NW = (size_t)DM * DM;  // 1048576
  unsigned short* base = (unsigned short*)d_ws;
  unsigned short* Wq = base + 0 * NW;   // all weights stored transposed [N][K]
  unsigned short* Wk = base + 1 * NW;
  unsigned short* Wv = base + 2 * NW;
  unsigned short* Wg = base + 3 * NW;
  unsigned short* Wo = base + 4 * NW;
  unsigned short* Qh = base + 5 * NW;   // 4*NW each (BB*SS*DM)
  unsigned short* Kh = base + 9 * NW;
  unsigned short* Vh = base + 13 * NW;  // head-transposed [b,h,d,s]
  unsigned short* Ao = base + 17 * NW;
  unsigned short* Gt = Qh;  // Q dead after attention -> reuse for gated output

  const int cb = (int)(NW / 1024);  // 256 thr * 4 elems
  cvtT_k<<<cb, 256, 0, stream>>>(wq, Wq);
  cvtT_k<<<cb, 256, 0, stream>>>(wk, Wk);
  cvtT_k<<<cb, 256, 0, stream>>>(wv, Wv);
  cvtT_k<<<cb, 256, 0, stream>>>(wg, Wg);
  cvtT_k<<<cb, 256, 0, stream>>>(wo, Wo);

  dim3 gg(64, 8);  // M/64 x N/128
  gemm_k<0><<<gg, 256, 0, stream>>>(q, nullptr, Wq, nullptr, nullptr, Qh, nullptr);
  gemm_k<0><<<gg, 256, 0, stream>>>(k, nullptr, Wk, nullptr, nullptr, Kh, nullptr);
  gemm_k<3><<<gg, 256, 0, stream>>>(v, nullptr, Wv, nullptr, nullptr, Vh, nullptr);

  attn_k<<<BB * NH * (SS / 64), 128, 0, stream>>>(Qh, Kh, Vh, swt, Ao);

  gemm_k<1><<<gg, 256, 0, stream>>>(nullptr, Ao, Wg, gb, Ao, Gt, nullptr);
  gemm_k<2><<<gg, 256, 0, stream>>>(nullptr, Gt, Wo, nullptr, nullptr, nullptr,
                                    (float*)d_out);
}
